// DistForecast_17343077941675
// MI455X (gfx1250) — matrix operand order repaired and checked by reading, not compiled, not run on hardware
//
#include <hip/hip_runtime.h>
#include <math.h>

// ---------------------------------------------------------------------------
// DistForecast on MI455X (gfx1250, wave32, WMMA bf16 16x16x32, f32 accum)
//
// Algebra: u[t] = x_t @ W_eff.T + b_eff, W_eff = W_hi @ W_in  (folds the input
// GEMM and it@W_hi.T out of the scan). Sequential part: h = tanh(u[t] + h@W_hh.T).
// GEMM: block = 4 waves = 64x64 tile; wave = 16x64 strip (4 accumulators,
// A-frag reused 4x); W-slab (32K x 64N, 4KB) double-buffered in LDS, staged
// with GLOBAL_LOAD_ASYNC_TO_LDS_B128 (ASYNCcnt) when the toolchain has it.
// ---------------------------------------------------------------------------

#define Bb 512
#define Nn 128
#define Tt 512
#define Hh 512

typedef __bf16 bf16;
typedef __attribute__((ext_vector_type(16))) bf16         v16bf;
typedef __attribute__((ext_vector_type(8)))  float        v8f;
typedef __attribute__((ext_vector_type(4)))  unsigned int v4u;
typedef __attribute__((ext_vector_type(4)))  int          v4i;

#if defined(__gfx1250__) && __has_builtin(__builtin_amdgcn_global_load_async_to_lds_b128) && \
    __has_builtin(__builtin_amdgcn_s_wait_asynccnt)
#define HAVE_ASYNC_LDS 1
typedef __attribute__((address_space(1))) v4i* gv4i_p;   // global int4*
typedef __attribute__((address_space(3))) v4i* lv4i_p;   // LDS int4*
#else
#define HAVE_ASYNC_LDS 0
#endif

union FragU { v16bf v; v4u u[2]; };

// 16B global -> LDS copy (async path bypasses VGPRs, tracked by ASYNCcnt)
__device__ __forceinline__ void copy16_g2l(const bf16* g, bf16* l) {
#if HAVE_ASYNC_LDS
  __builtin_amdgcn_global_load_async_to_lds_b128((gv4i_p)(g), (lv4i_p)(l), 0, 0);
#else
  *(v4u*)l = *(const v4u*)g;
#endif
}

__device__ __forceinline__ void wait_async_lds() {
#if HAVE_ASYNC_LDS
  __builtin_amdgcn_s_wait_asynccnt(0);
#endif
}

// A fragment (16x32 bf16, row-major, ld elems/row). ISA 7.12.2:
// lanes 0-15 hold K=[0..7]U[16..23]; lanes 16-31 hold K=[8..15]U[24..31].
__device__ __forceinline__ v16bf frag_a(const bf16* __restrict__ A, int ld,
                                        int mbase, int kbase, int lane) {
  const bf16* p = A + (size_t)(mbase + (lane & 15)) * ld + kbase + ((lane >> 4) << 3);
  FragU f;
  f.u[0] = *(const v4u*)(p);        // K = khalf + 0..7
  f.u[1] = *(const v4u*)(p + 16);   // K = khalf + 16..23
  return f.v;
}

// B fragment for C = A @ W.T from a row-major W-slab (B[k,n] = W[jbase+n, k]).
// Lane l holds column n = l%16, K = (l/16)*8 + [0..7]U[16..23].
__device__ __forceinline__ v16bf frag_bT(const bf16* W, int ld,
                                         int jbase, int lane) {
  const bf16* p = W + (size_t)(jbase + (lane & 15)) * ld + ((lane >> 4) << 3);
  FragU f;
  f.u[0] = *(const v4u*)(p);
  f.u[1] = *(const v4u*)(p + 16);
  return f.v;
}

// ------------------------- precompute / conversion -------------------------

// x (B,N,T) f32 -> xbf[(t*B+b)*N + n] bf16, via LDS 32x32 tile transpose.
__global__ void k_conv_x(const float* __restrict__ x, bf16* __restrict__ xbf) {
  __shared__ bf16 tile[32][33];
  int b  = blockIdx.z;
  int t0 = blockIdx.x << 5;
  int n0 = blockIdx.y << 5;
  int tx = threadIdx.x & 31;        // t (read) / n (write)
  int ty = threadIdx.x >> 5;        // 8 rows per pass
#pragma unroll
  for (int i = 0; i < 32; i += 8)   // coalesced read over t
    tile[ty + i][tx] = (bf16)x[((size_t)b * Nn + n0 + ty + i) * Tt + t0 + tx];
  __syncthreads();
#pragma unroll
  for (int i = 0; i < 32; i += 8) { // write: 64B runs over n
    int t = t0 + ty + i;
    xbf[((size_t)t * Bb + b) * Nn + n0 + tx] = tile[tx][ty + i];
  }
}

// strided f32 -> bf16 (extract W_hh from W_h, convert W_hid / W_fc1)
__global__ void k_conv_w(const float* __restrict__ src, bf16* __restrict__ dst,
                         int rows, int cols, int ld, int off) {
  int i = blockIdx.x * blockDim.x + threadIdx.x;
  if (i >= rows * cols) return;
  int r = i / cols, c = i - r * cols;
  dst[i] = (bf16)src[(size_t)r * ld + off + c];
}

// W_eff[j,n] = sum_h W_h[j,h] * W_in[h,n]   (W_hi = W_h[:, :H])
__global__ void k_weff(const float* __restrict__ W_h, const float* __restrict__ W_in,
                       bf16* __restrict__ Weff) {
  int i = blockIdx.x * blockDim.x + threadIdx.x;   // H*N
  int j = i >> 7, n = i & 127;
  float acc = 0.f;
  for (int h = 0; h < Hh; ++h)
    acc += W_h[(size_t)j * (2 * Hh) + h] * W_in[(size_t)h * Nn + n];
  Weff[i] = (bf16)acc;
}

// b_eff[j] = b_h[j] + sum_h W_h[j,h] * b_in[h]
__global__ void k_beff(const float* __restrict__ W_h, const float* __restrict__ b_in,
                       const float* __restrict__ b_h, float* __restrict__ beff) {
  int j = blockIdx.x * blockDim.x + threadIdx.x;
  if (j >= Hh) return;
  float acc = b_h[j];
  for (int h = 0; h < Hh; ++h) acc += W_h[(size_t)j * (2 * Hh) + h] * b_in[h];
  beff[j] = acc;
}

// --------------------------- WMMA GEMM kernel ------------------------------
// MODE 0: c=bias, store bf16          (u precompute)
// MODE 1: c=f32(uinit), tanh, bf16 (+f32 if out_f32)   (RNN step)
// MODE 2: c=bias, relu, store bf16    (hidden layer)
// MODE 3: c=bias, store f32           (fc head)
template <int MODE>
__global__ void k_gemm(const bf16* __restrict__ A, int lda,
                       const bf16* __restrict__ W, int ldw, int K, int nblk,
                       const float* __restrict__ biasv,
                       const bf16* __restrict__ uinit,
                       bf16* __restrict__ out_bf, float* __restrict__ out_f32,
                       int ldo) {
  __shared__ __align__(16) bf16 Bs[2][64 * 32];  // double-buffered 32Kx64N slabs
  int tid  = threadIdx.x;                        // 128 threads = 4 waves
  int lane = tid & 31;
  int wave = tid >> 5;
  int jb = blockIdx.x % nblk, mb = blockIdx.x / nblk;
  int jblock = jb << 6;
  int mstrip = (mb << 6) + (wave << 4);
  int rowadd = (lane >> 4) << 3;
  int colb   = jblock + (lane & 15);

  v8f c[4];
  if (MODE == 1) {
#pragma unroll
    for (int s = 0; s < 4; ++s)
#pragma unroll
      for (int r = 0; r < 8; ++r)
        c[s][r] = (float)uinit[(size_t)(mstrip + rowadd + r) * ldo + colb + 16 * s];
  } else {
#pragma unroll
    for (int s = 0; s < 4; ++s) {
      float bi = biasv[colb + 16 * s];
#pragma unroll
      for (int r = 0; r < 8; ++r) c[s][r] = bi;
    }
  }

  // staging assignment: thread -> 32 contiguous bytes of the 64x32 slab
  int sn = tid >> 1;                 // 0..63  (row n of slab)
  int sk = (tid & 1) << 4;           // 0 / 16 (col k of slab)
  const bf16* wsrc = W + (size_t)(jblock + sn) * ldw + sk;
  int soff = (sn << 5) + sk;

  // prologue: stage chunk 0 into buffer 0
  copy16_g2l(wsrc, &Bs[0][soff]);
  copy16_g2l(wsrc + 8, &Bs[0][soff + 8]);

  int nchunks = K >> 5;
#pragma unroll 2
  for (int i = 0; i < nchunks; ++i) {
    int k0  = i << 5;
    int buf = i & 1;
    wait_async_lds();                // this wave's slab loads for chunk i done
    __syncthreads();                 // whole block's chunk-i slab visible;
                                     // everyone done reading buf^1 -> reusable
    if (i + 1 < nchunks) {           // stage chunk i+1, overlapped with WMMAs
      const bf16* src = wsrc + k0 + 32;
      bf16* sd = &Bs[buf ^ 1][soff];
      copy16_g2l(src, sd);
      copy16_g2l(src + 8, sd + 8);
      __builtin_prefetch(A + (size_t)(mstrip + (lane & 15)) * lda + k0 + 32, 0, 0);
    }
    v16bf a = frag_a(A, lda, mstrip, k0, lane);
    const bf16* bbase = &Bs[buf][0];
#pragma unroll
    for (int s = 0; s < 4; ++s) {
      v16bf b = frag_bT(bbase, 32, 16 * s, lane);
      c[s] = __builtin_amdgcn_wmma_f32_16x16x32_bf16(false, a, false, b,
                                                     (short)0, c[s], false, false);
    }
  }

#pragma unroll
  for (int s = 0; s < 4; ++s) {
    int coln = colb + 16 * s;
#pragma unroll
    for (int r = 0; r < 8; ++r) {
      size_t idx = (size_t)(mstrip + rowadd + r) * ldo + coln;
      float v = c[s][r];
      if (MODE == 1) {
        v = tanhf(v);
        out_bf[idx] = (bf16)v;
        if (out_f32) out_f32[idx] = v;
      } else if (MODE == 2) {
        out_bf[idx] = (bf16)fmaxf(v, 0.f);
      } else if (MODE == 3) {
        out_f32[idx] = v;
      } else {
        out_bf[idx] = (bf16)v;
      }
    }
  }
}

// ------------------------------ BN + tail ----------------------------------

__global__ void k_bn_stats(const float* __restrict__ h, float* __restrict__ mean,
                           float* __restrict__ var) {
  int j = blockIdx.x;                 // one block per column (H)
  int tid = threadIdx.x;              // 256 threads
  float s = 0.f, s2 = 0.f;
  for (int b = tid; b < Bb; b += 256) {
    float v = h[(size_t)b * Hh + j];
    s += v; s2 += v * v;
  }
  __shared__ float sm1[256], sm2[256];
  sm1[tid] = s; sm2[tid] = s2;
  __syncthreads();
  for (int off = 128; off > 0; off >>= 1) {
    if (tid < off) { sm1[tid] += sm1[tid + off]; sm2[tid] += sm2[tid + off]; }
    __syncthreads();
  }
  if (tid == 0) {
    float m = sm1[0] * (1.f / Bb);
    mean[j] = m;
    var[j] = sm2[0] * (1.f / Bb) - m * m;
  }
}

__global__ void k_bn_norm(const float* __restrict__ h, const float* __restrict__ mean,
                          const float* __restrict__ var, const float* __restrict__ gamma,
                          const float* __restrict__ beta, bf16* __restrict__ hn) {
  int i = blockIdx.x * blockDim.x + threadIdx.x;   // B*H
  int j = i & (Hh - 1);
  float v = (h[i] - mean[j]) * rsqrtf(var[j] + 1e-5f) * gamma[j] + beta[j];
  hn[i] = (bf16)v;
}

// split/relu/exp/sample; d_out = [sample | mu_r | logvar] each (B,N) f32
__global__ void k_final(const float* __restrict__ out2, const float* __restrict__ eps,
                        float* __restrict__ out) {
  int i = blockIdx.x * blockDim.x + threadIdx.x;   // B*N
  int b = i >> 7, n = i & 127;
  float mu     = out2[b * (2 * Nn) + n];
  float logvar = out2[b * (2 * Nn) + Nn + n];
  float mur    = fmaxf(mu, 0.f);
  float sample = eps[i] * (0.5f * expf(logvar)) + mur;
  out[i]               = sample;
  out[Bb * Nn + i]     = mur;
  out[2 * Bb * Nn + i] = logvar;
}

// ------------------------------- launcher ----------------------------------

extern "C" void kernel_launch(void* const* d_in, const int* in_sizes, int n_in,
                              void* d_out, int out_size, void* d_ws, size_t ws_size,
                              hipStream_t stream) {
  const float* x     = (const float*)d_in[0];
  const float* eps   = (const float*)d_in[1];
  const float* W_in  = (const float*)d_in[2];
  const float* b_in  = (const float*)d_in[3];
  const float* W_h   = (const float*)d_in[4];
  const float* b_h   = (const float*)d_in[5];
  const float* gamma = (const float*)d_in[6];
  const float* beta  = (const float*)d_in[7];
  const float* W_hid = (const float*)d_in[8];
  const float* b_hid = (const float*)d_in[9];
  const float* W_fc1 = (const float*)d_in[10];
  const float* b_fc1 = (const float*)d_in[11];

  char* p = (char*)d_ws;
  auto take = [&](size_t bytes) -> char* {
    char* r = p; p += (bytes + 255) & ~(size_t)255; return r;
  };
  bf16*  xbf  = (bf16*) take((size_t)Tt * Bb * Nn * 2);   //  64 MB
  bf16*  u    = (bf16*) take((size_t)Tt * Bb * Hh * 2);   // 256 MB
  bf16*  Weff = (bf16*) take((size_t)Hh * Nn * 2);
  float* beff = (float*)take((size_t)Hh * 4);
  bf16*  Whh  = (bf16*) take((size_t)Hh * Hh * 2);
  bf16*  Whid = (bf16*) take((size_t)Hh * Hh * 2);
  bf16*  Wfc1 = (bf16*) take((size_t)2 * Nn * Hh * 2);
  bf16*  h0   = (bf16*) take((size_t)Bb * Hh * 2);
  bf16*  h1   = (bf16*) take((size_t)Bb * Hh * 2);
  float* hfin = (float*)take((size_t)Bb * Hh * 4);
  float* mean = (float*)take((size_t)Hh * 4);
  float* var  = (float*)take((size_t)Hh * 4);
  bf16*  hnrm = (bf16*) take((size_t)Bb * Hh * 2);
  bf16*  ob   = (bf16*) take((size_t)Bb * Hh * 2);
  float* out2 = (float*)take((size_t)Bb * 2 * Nn * 4);

  (void)hipMemsetAsync(h0, 0, (size_t)Bb * Hh * 2, stream);   // h0 = zeros

  k_conv_x<<<dim3(Tt / 32, Nn / 32, Bb), 256, 0, stream>>>(x, xbf);
  k_conv_w<<<(Hh * Hh) / 256, 256, 0, stream>>>(W_h, Whh, Hh, Hh, 2 * Hh, Hh);
  k_conv_w<<<(Hh * Hh) / 256, 256, 0, stream>>>(W_hid, Whid, Hh, Hh, Hh, 0);
  k_conv_w<<<(2 * Nn * Hh) / 256, 256, 0, stream>>>(W_fc1, Wfc1, 2 * Nn, Hh, Hh, 0);
  k_weff<<<(Hh * Nn) / 256, 256, 0, stream>>>(W_h, W_in, Weff);
  k_beff<<<(Hh + 255) / 256, 256, 0, stream>>>(W_h, b_in, b_h, beff);

  // u = xbf @ W_eff.T + b_eff : M = T*B = 262144, N = 512, K = 128
  k_gemm<0><<<(Tt * Bb / 64) * (Hh / 64), 128, 0, stream>>>(
      xbf, Nn, Weff, Nn, Nn, Hh / 64, beff, nullptr, u, nullptr, Hh);

  // Sequential scan: 512 step GEMMs (B=512, H=512, K=512), 64 blocks each
  bf16* hp = h0; bf16* hn = h1;
  for (int t = 0; t < Tt; ++t) {
    k_gemm<1><<<(Bb / 64) * (Hh / 64), 128, 0, stream>>>(
        hp, Hh, Whh, Hh, Hh, Hh / 64, nullptr, u + (size_t)t * Bb * Hh,
        hn, (t == Tt - 1) ? hfin : (float*)nullptr, Hh);
    bf16* tmp = hp; hp = hn; hn = tmp;
  }

  k_bn_stats<<<Hh, 256, 0, stream>>>(hfin, mean, var);
  k_bn_norm<<<(Bb * Hh) / 256, 256, 0, stream>>>(hfin, mean, var, gamma, beta, hnrm);
  k_gemm<2><<<(Bb / 64) * (Hh / 64), 128, 0, stream>>>(
      hnrm, Hh, Whid, Hh, Hh, Hh / 64, b_hid, nullptr, ob, nullptr, Hh);
  k_gemm<3><<<(Bb / 64) * (2 * Nn / 64), 128, 0, stream>>>(
      ob, Hh, Wfc1, Hh, Hh, 2 * Nn / 64, b_fc1, nullptr, nullptr, out2, 2 * Nn);
  k_final<<<(Bb * Nn) / 256, 256, 0, stream>>>(out2, eps, (float*)d_out);
}
